// Points_SamplerV2_5892695130398
// MI455X (gfx1250) — matrix operand order repaired and checked by reading, not compiled, not run on hardware
//
#include <hip/hip_runtime.h>

// ============================================================================
// MI455X (gfx1250, wave32) implementation.
//
// Roofline: ~175 GFLOP total (decoder 3x3 conv 256->128 dominates), but only
// a few hundred MB of activation traffic at bf16 -> memory-bound at 23.3 TB/s
// (~25us). Strategy: bf16 activations, all convs as implicit GEMM on
// v_wmma_f32_16x16x32_bf16, BN/ReLU/bias fused into epilogues, and the final
// 1x1 conv + channel-mean fused into the decoder epilogue so the 134MB (f32) d1
// tensor is never written. Weights are pre-repacked into per-lane WMMA
// A-fragment images in contiguous k order (k = 16*(l/16)+j, as B) so each lane loads one contiguous
// 32B chunk; activations are pixel-major [p][c] so B fragments are contiguous
// 32B LDS reads.
// ============================================================================

typedef __attribute__((ext_vector_type(16))) __bf16 v16bf;
typedef __attribute__((ext_vector_type(8)))  float  v8f;

#define BN_EPS 1e-5f
#define NB   64      // batch
#define NPIX 4096    // 64x64
#define NPOINT 1024

static __device__ __forceinline__ unsigned short f2bf(float f) {
  // round-to-nearest-even f32 -> bf16 (bit manipulation; no __bf16 scalar ops)
  unsigned u = __float_as_uint(f);
  unsigned r = u + 0x7FFFu + ((u >> 16) & 1u);
  return (unsigned short)(r >> 16);
}

static __device__ __forceinline__ v8f wmma_bf16(v16bf a, v16bf b, v8f c) {
  return __builtin_amdgcn_wmma_f32_16x16x32_bf16(false, a, false, b, (short)0, c,
                                                 false, false);
}

// ---------------------------------------------------------------------------
// Weight repack kernels: emit bf16 A-fragments in register image order.
// A-fragment k order used here (B loads match): lane l -> row M=l%16; element j -> K =
// (l/16)*16 + j.
// ---------------------------------------------------------------------------
__global__ void k_repack_enc(const float* __restrict__ enc_w,
                             unsigned short* __restrict__ encA) {
  int t = blockIdx.x * blockDim.x + threadIdx.x;
  if (t >= 9 * 4 * 2 * 512) return;
  int j = t & 15, lane = (t >> 4) & 31, mt = (t >> 9) & 1, kb = (t >> 10) & 3,
      tap = t >> 12;
  int oc  = mt * 16 + (lane & 15);
  int cin = kb * 32 + ((lane >> 4) << 4) + j;
  encA[t] = f2bf(enc_w[(oc * 128 + cin) * 9 + tap]);  // enc_w (32,128,3,3)
}

__global__ void k_repack_dec(const float* __restrict__ dec_w,
                             unsigned short* __restrict__ decA) {
  int t = blockIdx.x * blockDim.x + threadIdx.x;
  if (t >= 2 * 9 * 4 * 8 * 512) return;
  int j = t & 15, lane = (t >> 4) & 31, mt = (t >> 9) & 7;
  int rest = t >> 12;            // = (half*9+tap)*4 + kb
  int kb = rest & 3; int q = rest >> 2;
  int tap = q % 9, half = q / 9; // half 0 = up channels, 1 = fc channels
  int oc  = mt * 16 + (lane & 15);
  int cin = half * 128 + kb * 32 + ((lane >> 4) << 4) + j;
  decA[t] = f2bf(dec_w[(oc * 256 + cin) * 9 + tap]);  // dec_w (128,256,3,3)
}

__global__ void k_repack_up(const float* __restrict__ up_w,
                            unsigned short* __restrict__ upA) {
  int t = blockIdx.x * blockDim.x + threadIdx.x;
  if (t >= 32 * 512) return;
  int j = t & 15, lane = (t >> 4) & 31, mt = t >> 9;
  int m  = mt * 16 + (lane & 15);           // row = (o, di, dj)
  int o = m >> 2, di = (m >> 1) & 1, dj = m & 1;
  int k = ((lane >> 4) << 4) + j;  // input channel
  upA[t] = f2bf(up_w[((k * 128 + o) * 2 + di) * 2 + dj]);  // up_w (32,128,2,2)
}

// ---------------------------------------------------------------------------
// K1: conv1d(3->64)+BN+ReLU fused with concat -> fc pixel-major bf16 [b][p][128]
// ---------------------------------------------------------------------------
__global__ void k_build_fc(const float* __restrict__ points,
                           const float* __restrict__ features,
                           const float* __restrict__ w1,
                           const float* __restrict__ b1,
                           const float* __restrict__ g, const float* __restrict__ be,
                           const float* __restrict__ m, const float* __restrict__ v,
                           unsigned short* __restrict__ fc_pm) {
  int t = blockIdx.x * blockDim.x + threadIdx.x;   // t = ((b*4096+p)*128+c)
  int c = t & 127, p = (t >> 7) & 4095, b = t >> 19;
  float val;
  if (c < 64) {
    const float* pt = points + ((size_t)(b * 4096 + p)) * 3;
    float x = b1[c] + pt[0] * w1[c * 3 + 0] + pt[1] * w1[c * 3 + 1] +
              pt[2] * w1[c * 3 + 2];
    float s = g[c] * rsqrtf(v[c] + BN_EPS);
    val = fmaxf((x - m[c]) * s + be[c], 0.f);
  } else {
    val = features[((size_t)(b * 64 + (c - 64))) * 4096 + p];
  }
  fc_pm[t] = f2bf(val);
}

// ---------------------------------------------------------------------------
// K2: encoder conv3x3 128->32 + bias + BN + ReLU, implicit GEMM via WMMA.
// 1 block (4 waves) per output row; each wave: 16 pixels x 32 out-channels.
// ---------------------------------------------------------------------------
__global__ void k_encoder(const unsigned short* __restrict__ fc_pm,
                          const unsigned short* __restrict__ encA,
                          const float* __restrict__ enc_b,
                          const float* __restrict__ g, const float* __restrict__ be,
                          const float* __restrict__ m, const float* __restrict__ v,
                          float* __restrict__ enc_out) {
  __shared__ alignas(64) unsigned short tile[3][66][128];  // 3 rows, halo cols
  __shared__ float esc[32], esh[32];
  int b = blockIdx.x >> 6, h = blockIdx.x & 63;
  int tid = threadIdx.x, lane = tid & 31, wv = tid >> 5;
  int ln = lane & 15, lg = lane >> 4, w0 = wv * 16;

  if (tid < 32) {
    float s = g[tid] * rsqrtf(v[tid] + BN_EPS);
    esc[tid] = s;
    esh[tid] = enc_b[tid] * s + be[tid] - m[tid] * s;  // fold conv bias into BN
  }
  unsigned int* tz = (unsigned int*)&tile[0][0][0];
  for (int i = tid; i < 3 * 66 * 128 / 2; i += 128) tz[i] = 0u;
  __syncthreads();
  for (int t = tid; t < 192; t += 128) {               // 3 rows x 64 cols
    int dy = t / 64, w = t - dy * 64, gy = h + dy - 1;
    if ((unsigned)gy < 64u) {
      const uint4* gsrc =
          (const uint4*)(fc_pm + ((size_t)(b * 4096 + gy * 64 + w)) * 128);
      uint4* dst = (uint4*)&tile[dy][w + 1][0];
#pragma unroll
      for (int q = 0; q < 16; ++q) dst[q] = gsrc[q];
    }
  }
  __syncthreads();

  v8f acc[2]; acc[0] = (v8f){}; acc[1] = (v8f){};
#pragma unroll
  for (int tap = 0; tap < 9; ++tap) {
    int dy = tap / 3, dx = tap % 3;
#pragma unroll
    for (int kb = 0; kb < 4; ++kb) {
      v16bf bf = *(const v16bf*)&tile[dy][w0 + ln + dx][kb * 32 + lg * 16];
      const unsigned short* ab = encA + tap * 4096 + kb * 1024 + lane * 16;
      v16bf a0 = *(const v16bf*)ab;
      v16bf a1 = *(const v16bf*)(ab + 512);
      acc[0] = wmma_bf16(a0, bf, acc[0]);
      acc[1] = wmma_bf16(a1, bf, acc[1]);
    }
  }

  int p = h * 64 + w0 + ln;
  float* outp = enc_out + ((size_t)(b * 4096 + p)) * 32;
#pragma unroll
  for (int mt = 0; mt < 2; ++mt) {
    int oc0 = mt * 16 + lg * 8;
    float4 o0, o1;
#pragma unroll
    for (int r = 0; r < 4; ++r) {
      (&o0.x)[r] = fmaxf(acc[mt][r] * esc[oc0 + r] + esh[oc0 + r], 0.f);
      (&o1.x)[r] = fmaxf(acc[mt][r + 4] * esc[oc0 + r + 4] + esh[oc0 + r + 4], 0.f);
    }
    *(float4*)(outp + oc0 + 0) = o0;
    *(float4*)(outp + oc0 + 4) = o1;
  }
}

// ---------------------------------------------------------------------------
// K3: 2x2 maxpool -> e1 pixel-major bf16 [b][1024][32]
// ---------------------------------------------------------------------------
__global__ void k_pool(const float* __restrict__ enc_out,
                       unsigned short* __restrict__ e1_pm) {
  int t = blockIdx.x * blockDim.x + threadIdx.x;   // ((b*1024+q)*32+c)
  int c = t & 31, q = (t >> 5) & 1023, b = t >> 15;
  int i = q >> 5, j = q & 31;
  const float* base = enc_out + ((size_t)b * 4096) * 32;
  float m0 = base[((2 * i) * 64 + 2 * j) * 32 + c];
  float m1 = base[((2 * i) * 64 + 2 * j + 1) * 32 + c];
  float m2 = base[((2 * i + 1) * 64 + 2 * j) * 32 + c];
  float m3 = base[((2 * i + 1) * 64 + 2 * j + 1) * 32 + c];
  e1_pm[t] = f2bf(fmaxf(fmaxf(m0, m1), fmaxf(m2, m3)));
}

// ---------------------------------------------------------------------------
// K4: ConvTranspose2d 32->128 k2 s2 as a single-K-step WMMA GEMM:
// M=(o,di,dj)=512 rows, N=1024 pixels, K=32. One wave per 16x16 tile.
// ---------------------------------------------------------------------------
__global__ void k_upconv(const unsigned short* __restrict__ e1_pm,
                         const unsigned short* __restrict__ upA,
                         const float* __restrict__ up_b,
                         unsigned short* __restrict__ up_pm) {
  int wid = blockIdx.x * 8 + (threadIdx.x >> 5);
  int b = wid >> 11, r2 = wid & 2047, mt = r2 >> 6, nt = r2 & 63;
  int lane = threadIdx.x & 31, ln = lane & 15, lg = lane >> 4;
  int p = nt * 16 + ln;
  v16bf bf = *(const v16bf*)(e1_pm + ((size_t)((b << 10) | p)) * 32 + lg * 16);
  v16bf af = *(const v16bf*)(upA + (mt * 32 + lane) * 16);
  v8f z = (v8f){};
  v8f acc = wmma_bf16(af, bf, z);
  int i = p >> 5, j = p & 31;
#pragma unroll
  for (int r = 0; r < 8; ++r) {
    int mrow = mt * 16 + lg * 8 + r;
    int o = mrow >> 2, di = (mrow >> 1) & 1, dj = mrow & 1;
    int po = (2 * i + di) * 64 + (2 * j + dj);
    up_pm[((size_t)((b << 12) | po)) * 128 + o] = f2bf(acc[r] + up_b[o]);
  }
}

// ---------------------------------------------------------------------------
// K5: decoder conv3x3 256->128 + bias + BN + ReLU, fused with 1x1 conv(128->2)
// and channel-mean -> per-pixel score. d1 is never materialized.
// 256 input channels processed as two 128-ch LDS passes (up half, fc half).
// ---------------------------------------------------------------------------
__global__ void k_decoder(const unsigned short* __restrict__ up_pm,
                          const unsigned short* __restrict__ fc_pm,
                          const unsigned short* __restrict__ decA,
                          const float* __restrict__ dec_b,
                          const float* __restrict__ g, const float* __restrict__ be,
                          const float* __restrict__ m, const float* __restrict__ v,
                          const float* __restrict__ last_w,
                          const float* __restrict__ last_b,
                          float* __restrict__ scores) {
  __shared__ alignas(64) unsigned short tile[3][66][128];
  __shared__ float dsc[128], dsh[128], dwb[128];
  int b = blockIdx.x >> 6, h = blockIdx.x & 63;
  int tid = threadIdx.x, lane = tid & 31, wv = tid >> 5;
  int ln = lane & 15, lg = lane >> 4, w0 = wv * 16;

  if (tid < 128) {
    float s = g[tid] * rsqrtf(v[tid] + BN_EPS);
    dsc[tid] = s;
    dsh[tid] = dec_b[tid] * s + be[tid] - m[tid] * s;
    dwb[tid] = 0.5f * (last_w[tid] + last_w[128 + tid]);  // mean of 2 out ch
  }

  v8f acc[8];
#pragma unroll
  for (int i = 0; i < 8; ++i) acc[i] = (v8f){};

  for (int half = 0; half < 2; ++half) {
    const unsigned short* src = (half == 0) ? up_pm : fc_pm;
    __syncthreads();
    unsigned int* tz = (unsigned int*)&tile[0][0][0];
    for (int i = tid; i < 3 * 66 * 128 / 2; i += 128) tz[i] = 0u;
    __syncthreads();
    for (int t = tid; t < 192; t += 128) {
      int dy = t / 64, w = t - dy * 64, gy = h + dy - 1;
      if ((unsigned)gy < 64u) {
        const uint4* gsrc =
            (const uint4*)(src + ((size_t)(b * 4096 + gy * 64 + w)) * 128);
        uint4* dst = (uint4*)&tile[dy][w + 1][0];
#pragma unroll
        for (int q = 0; q < 16; ++q) dst[q] = gsrc[q];
      }
    }
    __syncthreads();

#pragma unroll
    for (int tap = 0; tap < 9; ++tap) {
      int dy = tap / 3, dx = tap % 3;
      const unsigned short* tapbase =
          decA + (half * 9 + tap) * 16384 + lane * 16;
      if (tap < 8)  // warm GL2 for next tap's A fragments (global_prefetch_b8)
        __builtin_prefetch(tapbase + 16384, 0, 0);
#pragma unroll
      for (int kb = 0; kb < 4; ++kb) {
        v16bf bf = *(const v16bf*)&tile[dy][w0 + ln + dx][kb * 32 + lg * 16];
        const unsigned short* ab = tapbase + kb * 4096;
#pragma unroll
        for (int mt = 0; mt < 8; ++mt) {
          v16bf af = *(const v16bf*)(ab + mt * 512);
          acc[mt] = wmma_bf16(af, bf, acc[mt]);
        }
      }
    }
  }

  // Epilogue: BN+ReLU then dot with dwb; reduce the two half-wave oc groups.
  float part = 0.f;
#pragma unroll
  for (int mt = 0; mt < 8; ++mt) {
#pragma unroll
    for (int r = 0; r < 8; ++r) {
      int oc = mt * 16 + lg * 8 + r;
      float val = fmaxf(acc[mt][r] * dsc[oc] + dsh[oc], 0.f);
      part += dwb[oc] * val;
    }
  }
  part += __shfl_xor(part, 16, 32);  // lanes l and l+16 hold same pixel
  if (lg == 0)
    scores[(b << 12) + h * 64 + w0 + ln] = part + 0.5f * (last_b[0] + last_b[1]);
}

// ---------------------------------------------------------------------------
// K6: per-batch full bitonic sort (descending, index-ascending tie-break, to
// match jax.lax.top_k) of 4096 scores in LDS; emit top-1024 indices + preds.
// ---------------------------------------------------------------------------
__global__ void k_topk(const float* __restrict__ scores,
                       int* __restrict__ idx_out,
                       float* __restrict__ preds_out) {
  __shared__ float sv[4096];
  __shared__ int si[4096];
  int b = blockIdx.x, tid = threadIdx.x;
  for (int i = tid; i < 4096; i += 1024) {
    float val = scores[(b << 12) + i];
    sv[i] = val; si[i] = i;
    preds_out[(b << 12) + i] = val;   // preds_2 in original order
  }
  __syncthreads();
  for (int k = 2; k <= 4096; k <<= 1) {
    for (int j = k >> 1; j > 0; j >>= 1) {
      for (int i = tid; i < 4096; i += 1024) {
        int ixj = i ^ j;
        if (ixj > i) {
          bool desc = ((i & k) == 0);
          float va = sv[i], vb = sv[ixj];
          int ia = si[i], ib = si[ixj];
          bool a_gt_b = (va > vb) || (va == vb && ia < ib);
          bool doswap = desc ? !a_gt_b : a_gt_b;
          if (doswap) { sv[i] = vb; sv[ixj] = va; si[i] = ib; si[ixj] = ia; }
        }
      }
      __syncthreads();
    }
  }
  if (tid < NPOINT) idx_out[b * NPOINT + tid] = si[tid];
}

// ---------------------------------------------------------------------------
extern "C" void kernel_launch(void* const* d_in, const int* in_sizes, int n_in,
                              void* d_out, int out_size, void* d_ws, size_t ws_size,
                              hipStream_t stream) {
  const float* points   = (const float*)d_in[0];
  const float* features = (const float*)d_in[1];
  const float* conv1_w  = (const float*)d_in[2];
  const float* conv1_b  = (const float*)d_in[3];
  const float* bn1_g = (const float*)d_in[4];
  const float* bn1_b = (const float*)d_in[5];
  const float* bn1_m = (const float*)d_in[6];
  const float* bn1_v = (const float*)d_in[7];
  const float* enc_w = (const float*)d_in[8];
  const float* enc_b = (const float*)d_in[9];
  const float* ebn_g = (const float*)d_in[10];
  const float* ebn_b = (const float*)d_in[11];
  const float* ebn_m = (const float*)d_in[12];
  const float* ebn_v = (const float*)d_in[13];
  const float* up_w  = (const float*)d_in[14];
  const float* up_b  = (const float*)d_in[15];
  const float* dec_w = (const float*)d_in[16];
  const float* dec_b = (const float*)d_in[17];
  const float* dbn_g = (const float*)d_in[18];
  const float* dbn_b = (const float*)d_in[19];
  const float* dbn_m = (const float*)d_in[20];
  const float* dbn_v = (const float*)d_in[21];
  const float* last_w = (const float*)d_in[22];
  const float* last_b = (const float*)d_in[23];
  // d_in[24] = npoint (fixed 1024 per setup_inputs)

  // Workspace layout (256B aligned), ~174 MB total.
  char* ws = (char*)d_ws;
  size_t off = 0;
  auto take = [&](size_t bytes) {
    size_t cur = off;
    off += (bytes + 255) & ~(size_t)255;
    return (void*)(ws + cur);
  };
  unsigned short* fc_pm  = (unsigned short*)take((size_t)NB * NPIX * 128 * 2);
  unsigned short* up_pm  = (unsigned short*)take((size_t)NB * NPIX * 128 * 2);
  float*          enc_out= (float*)take((size_t)NB * NPIX * 32 * 4);
  unsigned short* e1_pm  = (unsigned short*)take((size_t)NB * 1024 * 32 * 2);
  unsigned short* encA   = (unsigned short*)take((size_t)9 * 4 * 2 * 512 * 2);
  unsigned short* decA   = (unsigned short*)take((size_t)2 * 9 * 4 * 8 * 512 * 2);
  unsigned short* upA    = (unsigned short*)take((size_t)32 * 512 * 2);
  float*          scores = (float*)take((size_t)NB * NPIX * 4);
  (void)ws_size;

  int* idx_out = (int*)d_out;                        // (B,1024) int32
  float* preds_out = (float*)d_out + NB * NPOINT;    // (B,4096,1) f32

  // Weight repacks (tiny, independent)
  k_repack_enc<<<(9 * 4 * 2 * 512 + 255) / 256, 256, 0, stream>>>(enc_w, encA);
  k_repack_dec<<<(2 * 9 * 4 * 8 * 512 + 255) / 256, 256, 0, stream>>>(dec_w, decA);
  k_repack_up<<<(32 * 512 + 255) / 256, 256, 0, stream>>>(up_w, upA);

  // K1: build fc (conv1d+BN+ReLU || features) pixel-major bf16
  k_build_fc<<<(NB * NPIX * 128) / 256, 256, 0, stream>>>(
      points, features, conv1_w, conv1_b, bn1_g, bn1_b, bn1_m, bn1_v, fc_pm);

  // K2: encoder conv (WMMA) ; K3: maxpool
  k_encoder<<<NB * 64, 128, 0, stream>>>(fc_pm, encA, enc_b, ebn_g, ebn_b,
                                         ebn_m, ebn_v, enc_out);
  k_pool<<<(NB * 1024 * 32) / 256, 256, 0, stream>>>(enc_out, e1_pm);

  // K4: transpose-conv (WMMA, single K step)
  k_upconv<<<(NB * 32 * 64) / 8, 256, 0, stream>>>(e1_pm, upA, up_b, up_pm);

  // K5: decoder conv (WMMA) fused with 1x1 conv + channel-mean -> scores
  k_decoder<<<NB * 64, 128, 0, stream>>>(up_pm, fc_pm, decA, dec_b, dbn_g,
                                         dbn_b, dbn_m, dbn_v, last_w, last_b,
                                         scores);

  // K6: per-batch top-1024 (bitonic sort) + preds output
  k_topk<<<NB, 1024, 0, stream>>>(scores, idx_out, preds_out);
}
